// Weight_83038897701252
// MI455X (gfx1250) — compile-verified, matrix operand order checked by reading, not run on hardware
//
#include <hip/hip_runtime.h>
#include <hip/hip_bf16.h>

typedef __attribute__((ext_vector_type(16))) __bf16 v16bf;
typedef __attribute__((ext_vector_type(8)))  __bf16 v8bf;
typedef __attribute__((ext_vector_type(8)))  float  v8f;

#define S_LEN 8192
#define D_DIM 4096
#define H_N   8
#define G_N   4
#define HD_N  128
#define EPSV  1e-6f

#define BLOCK 512     // 16 waves (wave32)
#define BM    64      // tokens per workgroup
#define NQ    512     // q columns per head (G*HD)
#define NK    128     // k columns per head
#define NTOT  640     // combined N
#define NTW   10      // N-tiles per wave
#define KSTEP 64      // K per staging step (2 wmma K-slices of 32)
#define AS    72      // A LDS row stride (bf16 elems, padded; 144B, 16B-aligned)
#define WS    72      // W LDS row stride (bf16 elems, padded)
#define CS    648     // C LDS row stride (f32 elems, padded)
#define A_BYTES (BM * AS * 2)          // 9216
#define SMEM_BYTES (BM * CS * 4)       // 165888 (> A_BYTES + NTOT*WS*2 = 101376)

__device__ __forceinline__ v8bf cvt8(const float* p) {
    float4 f0 = *(const float4*)(p);
    float4 f1 = *(const float4*)(p + 4);
    v8bf o;
    o[0] = (__bf16)f0.x; o[1] = (__bf16)f0.y; o[2] = (__bf16)f0.z; o[3] = (__bf16)f0.w;
    o[4] = (__bf16)f1.x; o[5] = (__bf16)f1.y; o[6] = (__bf16)f1.z; o[7] = (__bf16)f1.w;
    return o;
}

__device__ __forceinline__ v16bf frag16(const __bf16* p) {
    // lane's 16 bf16: chunk0 at +0, chunk1 at +16 elems (K layout per ISA)
    v8bf lo = *(const v8bf*)p;
    v8bf hi = *(const v8bf*)(p + 16);
    return __builtin_shufflevector(lo, hi, 0,1,2,3,4,5,6,7,8,9,10,11,12,13,14,15);
}

__global__ __launch_bounds__(BLOCK, 1)
void fused_gatescore_kernel(const float* __restrict__ hs,
                            const float* __restrict__ q_w,
                            const float* __restrict__ q_b,
                            const float* __restrict__ k_w,
                            const float* __restrict__ q_norm_w,
                            const float* __restrict__ k_norm_w,
                            const float* __restrict__ k_base,
                            const float* __restrict__ bvec,
                            float* __restrict__ out)
{
    extern __shared__ unsigned char smem[];
    __bf16* Ald = (__bf16*)smem;                 // [BM][AS] bf16 (GEMM stage)
    __bf16* Wld = (__bf16*)(smem + A_BYTES);     // [NTOT][WS] bf16 (GEMM stage)
    float*  Cld = (float*)smem;                  // [BM][CS] f32  (epilogue stage)

    const int tid  = threadIdx.x;
    const int h    = blockIdx.y;
    const int s0   = blockIdx.x * BM;
    const int lane = tid & 31;
    const int wv   = tid >> 5;          // wave id 0..15
    const int hlf  = lane >> 4;         // 0: lanes 0-15, 1: lanes 16-31
    const int l16  = lane & 15;
    const int mt   = wv & 3;            // M-tile (16 rows) this wave owns
    const int nbase = (wv >> 2) * 160;  // N-quarter this wave owns (10 tiles of 16)

    v8f acc[NTW] = {};

    // cooperative A-load mapping: 64 rows x 8 segments = 512 tasks, 1 per thread
    const int arow = tid >> 3;
    const int aseg = tid & 7;
    const float* aptr = hs + (size_t)(s0 + arow) * D_DIM + aseg * 8;

    for (int d0 = 0; d0 < D_DIM; d0 += KSTEP) {
        __syncthreads();

        // ---- stage A tile: 64 x 64 f32 -> bf16 LDS (1 segment per thread) ----
        *(v8bf*)(Ald + arow * AS + aseg * 8) = cvt8(aptr + d0);

        // ---- stage W tile: 640 rows x 64 f32 -> bf16 LDS (1-2 rows per thread) ----
        for (int r = tid; r < NTOT; r += BLOCK) {
            const float* wrow = (r < NQ)
                ? (q_w + (size_t)(h * NQ + r) * D_DIM + d0)
                : (k_w + (size_t)(h * NK + (r - NQ)) * D_DIM + d0);
            if (d0 + KSTEP < D_DIM) __builtin_prefetch(wrow + KSTEP, 0, 1);
            #pragma unroll
            for (int seg = 0; seg < 8; ++seg) {
                *(v8bf*)(Wld + r * WS + seg * 8) = cvt8(wrow + seg * 8);
            }
        }
        __syncthreads();

        // ---- two K=32 slices per staged tile ----
        #pragma unroll
        for (int kk = 0; kk < 2; ++kk) {
            const int koff = kk * 32;
            // A fragment (16x32): lanes 0-15 -> K 0..7,16..23; lanes 16-31 -> K 8..15,24..31
            const v16bf afrag = frag16(Ald + (mt * 16 + l16) * AS + koff + hlf * 8);

            // Preload ALL B fragments of this K-slice (independent loads -> one ds clause),
            // then issue WMMAs back-to-back with only partial dscnt waits.
            v16bf bf[NTW];
            #pragma unroll
            for (int nt = 0; nt < NTW; ++nt) {
                bf[nt] = frag16(Wld + (nbase + nt * 16 + l16) * WS + koff + hlf * 8);
            }
            #pragma unroll
            for (int nt = 0; nt < NTW; ++nt) {
                acc[nt] = __builtin_amdgcn_wmma_f32_16x16x32_bf16(
                    false, afrag, false, bf[nt], (short)0, acc[nt], false, false);
            }
        }
    }

    __syncthreads();
    // ---- dump accumulators to LDS C tile [BM][CS]; VGPR r -> M = 16*mt + 8*hlf + r, N = lane&15 ----
    #pragma unroll
    for (int nt = 0; nt < NTW; ++nt) {
        const int n = nbase + nt * 16 + l16;
        #pragma unroll
        for (int r = 0; r < 8; ++r) {
            Cld[(mt * 16 + hlf * 8 + r) * CS + n] = acc[nt][r];
        }
    }
    __syncthreads();

    // ---- fused epilogue: bias, RMS norms, dots, logistic sink score ----
    if (tid < BM) {
        const int m = tid;
        const float* crow = Cld + m * CS;              // [0..511]=q raw, [512..639]=k raw
        const float inv_scale = 0.08838834764831845f;  // 1/sqrt(HD)

        float ssk = 0.f;
        for (int d = 0; d < HD_N; ++d) { float kv = crow[NQ + d]; ssk += kv * kv; }
        const float rk = rsqrtf(ssk * (1.0f / HD_N) + EPSV);

        float score = 0.f;
        for (int g = 0; g < G_N; ++g) {
            float ssq = 0.f, dqk = 0.f, dbase = 0.f;
            for (int d = 0; d < HD_N; ++d) {
                float qv = crow[g * HD_N + d] + q_b[h * NQ + g * HD_N + d];
                ssq += qv * qv;
                float qn = qv * q_norm_w[d];
                dqk   += (crow[NQ + d] * k_norm_w[d]) * qn;
                dbase += k_base[h * HD_N + d] * qn;
            }
            const float rq    = rsqrtf(ssq * (1.0f / HD_N) + EPSV);
            const float logit = dqk * rk * rq * inv_scale + bvec[h * G_N + g];
            const float lbase = dbase * rq * inv_scale;
            score += 1.f / (1.f + expf(lbase - logit));
        }
        out[(size_t)h * S_LEN + (s0 + m)] = score * 0.25f;
    }
}

extern "C" void kernel_launch(void* const* d_in, const int* in_sizes, int n_in,
                              void* d_out, int out_size, void* d_ws, size_t ws_size,
                              hipStream_t stream) {
    const float* hs       = (const float*)d_in[0];
    const float* q_w      = (const float*)d_in[1];
    const float* q_b      = (const float*)d_in[2];
    const float* k_w      = (const float*)d_in[3];
    const float* q_norm_w = (const float*)d_in[4];
    const float* k_norm_w = (const float*)d_in[5];
    const float* k_base   = (const float*)d_in[6];
    const float* bvec     = (const float*)d_in[7];
    float* out = (float*)d_out;

    dim3 grid(S_LEN / BM, H_N);
    fused_gatescore_kernel<<<grid, BLOCK, SMEM_BYTES, stream>>>(
        hs, q_w, q_b, k_w, q_norm_w, k_norm_w, k_base, bvec, out);
}
